// TransformerBlock_78297253806494
// MI455X (gfx1250) — matrix operand order repaired and checked by reading, not compiled, not run on hardware
//
#include <hip/hip_runtime.h>

// ---- problem constants (match reference) ----
#define E_   1024
#define H_   16
#define DH_  64
#define B_   2
#define T_   2048
#define M_   (B_*T_)     // 4096 rows
#define FF_  (4*E_)      // 4096

// ---- vector types for WMMA ----
typedef _Float16 v8h  __attribute__((ext_vector_type(8)));
typedef _Float16 v16h __attribute__((ext_vector_type(16)));
typedef float    v8f  __attribute__((ext_vector_type(8)));
typedef int      v4i  __attribute__((ext_vector_type(4)));

// ---- CDNA5 async global->LDS path (probe-guarded; fallback = direct stores) ----
#if defined(__AMDGCN__) && __has_builtin(__builtin_amdgcn_global_load_async_to_lds_b128)
#define HAVE_ASYNC_LDS 1
typedef __attribute__((address_space(1))) v4i* gv4i_p;   // prints as "int4 __device__ *"
typedef __attribute__((address_space(3))) v4i* lv4i_p;   // prints as "int4 __shared__ *"
#else
#define HAVE_ASYNC_LDS 0
#endif

__device__ __forceinline__ void wait_async0() {
#if defined(__AMDGCN__)
#if __has_builtin(__builtin_amdgcn_s_wait_asynccnt)
    __builtin_amdgcn_s_wait_asynccnt(0);
#else
    asm volatile("s_wait_asynccnt 0" ::: "memory");
#endif
#endif
}

__device__ __forceinline__ v16h cat8(v8h lo, v8h hi) {
    return __builtin_shufflevector(lo, hi, 0,1,2,3,4,5,6,7,8,9,10,11,12,13,14,15);
}
__device__ __forceinline__ v16h ld16(const _Float16* p) {
    v8h lo = *(const v8h*)p;
    v8h hi = *(const v8h*)(p + 8);
    return cat8(lo, hi);
}
__device__ __forceinline__ v8f wmma16(v16h a, v16h b, v8f c) {
    // D = A(16x32 f16) * B(32x16 f16) + C(16x16 f32)
    return __builtin_amdgcn_wmma_f32_16x16x32_f16(false, a, false, b, (short)0, c, false, false);
}
__device__ __forceinline__ v8f vzero8() { v8f z; for (int i = 0; i < 8; ++i) z[i] = 0.f; return z; }

__device__ __forceinline__ float gelu_ref(float x) {
    // faithful: 0.5*x*(1+tanh(sqrt(2/pi)*x + 0.044715*x^3))
    return 0.5f * x * (1.f + tanhf(0.7978845608028654f * x + 0.044715f * x * x * x));
}

// =====================================================================
// Weight conversion kernels (fp32 -> f16, one-time per launch)
// =====================================================================
__global__ __launch_bounds__(256) void pack_qkv_f16(const float* __restrict__ w,
                                                    _Float16* __restrict__ o) {
    // w: (H, E, DH) -> o: [E][H*DH]  (so Q = xn @ o is a plain GEMM)
    int idx = blockIdx.x * 256 + threadIdx.x;           // H*E*DH = 1M entries
    int d = idx & (DH_ - 1);
    int k = (idx >> 6) & (E_ - 1);
    int h = idx >> 16;                                  // idx / (E_*DH_)
    o[(size_t)k * E_ + h * DH_ + d] = (_Float16)w[idx];
}

__global__ __launch_bounds__(256) void cvt_f16(const float* __restrict__ w,
                                               _Float16* __restrict__ o, int n) {
    int idx = blockIdx.x * 256 + threadIdx.x;
    if (idx < n) o[idx] = (_Float16)w[idx];
}

// =====================================================================
// LayerNorm: one block per row of E_=1024, fp32 in -> f16 out
// =====================================================================
__global__ __launch_bounds__(256) void layernorm_f16(const float* __restrict__ x,
                                                     const float* __restrict__ sc,
                                                     const float* __restrict__ sh,
                                                     _Float16* __restrict__ out) {
    __shared__ float red[256];
    const int tid = threadIdx.x;
    const float* xr = x + (size_t)blockIdx.x * E_;
    float s = 0.f;
    for (int i = tid; i < E_; i += 256) s += xr[i];
    red[tid] = s; __syncthreads();
    for (int st = 128; st > 0; st >>= 1) { if (tid < st) red[tid] += red[tid + st]; __syncthreads(); }
    const float mean = red[0] * (1.f / E_);
    __syncthreads();
    float v = 0.f;
    for (int i = tid; i < E_; i += 256) { float d = xr[i] - mean; v += d * d; }
    red[tid] = v; __syncthreads();
    for (int st = 128; st > 0; st >>= 1) { if (tid < st) red[tid] += red[tid + st]; __syncthreads(); }
    const float rstd = rsqrtf(red[0] * (1.f / E_) + 1e-5f);
    _Float16* orow = out + (size_t)blockIdx.x * E_;
    for (int i = tid; i < E_; i += 256)
        orow[i] = (_Float16)(sc[i] * (xr[i] - mean) * rstd + sh[i]);
}

// =====================================================================
// Generic WMMA GEMM: C[M,N] = A[M,K](f16) * B[K,N](f16) (+bias)(gelu)(+res)
// 256 threads = 8 waves (2x4), block tile 128x128, k-step 64 (2 WMMA k-steps,
// 16 WMMAs per staging phase). A tile staged via async global->LDS when
// available; B tile transposed into LDS so B-fragments are contiguous.
// =====================================================================
#define BM 128
#define BN 128
#define BK 64
#define LP 8   // LDS row pad (halfs) -> 16B-aligned rows

template<bool BIAS, bool RES, bool GELU, bool OUTF16>
__global__ __launch_bounds__(256)
void gemm_wmma(const _Float16* __restrict__ A, const _Float16* __restrict__ Bw,
               const float* __restrict__ bias, const float* __restrict__ res,
               float* __restrict__ Cf, _Float16* __restrict__ Ch,
               int M, int N, int K) {
    __shared__ _Float16 As[BM][BK + LP];       // row-major A tile   (rows 144B, 16B aligned)
    __shared__ _Float16 Bs[BN][BK + LP];       // transposed B tile: Bs[n][k]

    const int tid  = threadIdx.x;
    const int row0 = blockIdx.y * BM;
    const int col0 = blockIdx.x * BN;
    const int wave = tid >> 5, lane = tid & 31;
    const int wm = wave & 1, wn = wave >> 1;   // 2 waves in M, 4 in N
    const int l16 = lane & 15, hs = lane >> 4;

    v8f acc[4][2];
    for (int mi = 0; mi < 4; ++mi) for (int ni = 0; ni < 2; ++ni) acc[mi][ni] = vzero8();

    for (int k0 = 0; k0 < K; k0 += BK) {
        // --- stage A: 128x64 halfs; 32 halfs (4x b128) per thread ---
        {
            int r  = tid >> 1;
            int kc = (tid & 1) << 5;
            const _Float16* src = A + (size_t)(row0 + r) * K + k0 + kc;
            _Float16* dst = &As[r][kc];
#if HAVE_ASYNC_LDS
            #pragma unroll
            for (int i = 0; i < 4; ++i)
                __builtin_amdgcn_global_load_async_to_lds_b128(
                    (gv4i_p)(src + i * 8), (lv4i_p)(dst + i * 8), 0, 0);
#else
            #pragma unroll
            for (int i = 0; i < 4; ++i)
                *(v8h*)(dst + i * 8) = *(const v8h*)(src + i * 8);
#endif
        }
        // --- stage B transposed: Bs[n][k] = B[k0+k][col0+n]; 32 halfs per thread ---
        {
            int kk = tid >> 2;                 // 0..63
            int nc = (tid & 3) << 5;           // 0,32,64,96
            const _Float16* src = Bw + (size_t)(k0 + kk) * N + col0 + nc;
            if (k0 + BK < K) __builtin_prefetch(src + (size_t)BK * N, 0, 1);  // global_prefetch
            #pragma unroll
            for (int i = 0; i < 32; ++i) Bs[nc + i][kk] = src[i];
        }
#if HAVE_ASYNC_LDS
        wait_async0();
#endif
        __syncthreads();

        // --- two WMMA k-steps per staging phase ---
        #pragma unroll
        for (int ks2 = 0; ks2 < 2; ++ks2) {
            const int kb = ks2 * 32;
            v16h af[4], bf[2];
            #pragma unroll
            for (int mi = 0; mi < 4; ++mi) {
                const _Float16* p = &As[wm * 64 + mi * 16 + l16][kb];
                af[mi] = cat8(*(const v8h*)(p + hs * 8), *(const v8h*)(p + 16 + hs * 8));
            }
            #pragma unroll
            for (int ni = 0; ni < 2; ++ni) {
                const _Float16* p = &Bs[wn * 32 + ni * 16 + l16][kb];
                bf[ni] = cat8(*(const v8h*)(p + hs * 8), *(const v8h*)(p + 16 + hs * 8));
            }
            #pragma unroll
            for (int mi = 0; mi < 4; ++mi)
                #pragma unroll
                for (int ni = 0; ni < 2; ++ni)
                    acc[mi][ni] = wmma16(af[mi], bf[ni], acc[mi][ni]);
        }
        __syncthreads();
    }

    // --- epilogue: C layout row = j + 8*hs (per reg), col = lane%16 ---
    #pragma unroll
    for (int mi = 0; mi < 4; ++mi) {
        #pragma unroll
        for (int ni = 0; ni < 2; ++ni) {
            #pragma unroll
            for (int j = 0; j < 8; ++j) {
                int r = row0 + wm * 64 + mi * 16 + j + hs * 8;
                int c = col0 + wn * 32 + ni * 16 + l16;
                float v = acc[mi][ni][j];
                if (BIAS) v += bias[c];
                if (GELU) v = gelu_ref(v);
                if (RES)  v += res[(size_t)r * N + c];
                if (OUTF16) Ch[(size_t)r * N + c] = (_Float16)v;
                else        Cf[(size_t)r * N + c] = v;
            }
        }
    }
}

// =====================================================================
// Flash attention (causal). Q,K,V are f16 [B,T,E] (head = col slice).
// One wave per 16-row query tile per (b,h); KV in 32-column chunks.
// Output scatter implements the reference's swapaxes/reshape quirk:
//   concat[b, d*32 + t/64, h*64 + t%64] = ctx[b,h,t,d]
// =====================================================================
__global__ __launch_bounds__(128)
void flash_attn(const _Float16* __restrict__ Q, const _Float16* __restrict__ Kv,
                const _Float16* __restrict__ V, _Float16* __restrict__ O) {
    __shared__ _Float16 Plds[4][16][32];       // per-wave P staging (f16, A layout source)

    const int bh   = blockIdx.x;
    const int b    = bh >> 4, h = bh & 15;
    const int wave = threadIdx.x >> 5, lane = threadIdx.x & 31;
    const int l16  = lane & 15, hs = lane >> 4;
    const int q0   = (blockIdx.y * 4 + wave) * 16;

    // Q fragments for d0 = 0, 32 (K dim of QK^T GEMM is the head dim, DH=64)
    v16h qf[2];
    #pragma unroll
    for (int ks = 0; ks < 2; ++ks) {
        const _Float16* qr = Q + ((size_t)b * T_ + (q0 + l16)) * E_ + h * DH_ + ks * 32;
        qf[ks] = cat8(*(const v8h*)(qr + hs * 16), *(const v8h*)(qr + hs * 16 + 8));
    }

    v8f acco[4]; for (int nn = 0; nn < 4; ++nn) acco[nn] = vzero8();
    float m_[8], l_[8];
    #pragma unroll
    for (int j = 0; j < 8; ++j) { m_[j] = -1e30f; l_[j] = 0.f; }

    for (int s0 = 0; s0 < q0 + 16; s0 += 32) {
        // ---- scores: two 16x16 tiles covering cols [s0, s0+32) ----
        v8f sc[2]; sc[0] = vzero8(); sc[1] = vzero8();
        #pragma unroll
        for (int nh = 0; nh < 2; ++nh) {
            int s = s0 + nh * 16 + l16;
            #pragma unroll
            for (int ks = 0; ks < 2; ++ks) {
                const _Float16* kr = Kv + ((size_t)b * T_ + s) * E_ + h * DH_ + ks * 32 + hs * 16;
                sc[nh] = wmma16(qf[ks], ld16(kr), sc[nh]);
            }
        }
        // ---- online softmax (rows live in accumulator regs; cols across 16 lanes) ----
        #pragma unroll
        for (int j = 0; j < 8; ++j) {
            const int row = q0 + j + hs * 8;
            float s0v = sc[0][j] * 0.125f; if (s0 + l16 > row)      s0v = -1e30f;
            float s1v = sc[1][j] * 0.125f; if (s0 + 16 + l16 > row) s1v = -1e30f;
            float mx = fmaxf(m_[j], fmaxf(s0v, s1v));
            #pragma unroll
            for (int o = 1; o < 16; o <<= 1) mx = fmaxf(mx, __shfl_xor(mx, o, 16));
            const float p0 = __expf(s0v - mx), p1 = __expf(s1v - mx);
            float rs = p0 + p1;
            #pragma unroll
            for (int o = 1; o < 16; o <<= 1) rs += __shfl_xor(rs, o, 16);
            const float f = __expf(m_[j] - mx);
            l_[j] = l_[j] * f + rs;
            m_[j] = mx;
            #pragma unroll
            for (int nn = 0; nn < 4; ++nn) acco[nn][j] *= f;
            Plds[wave][j + hs * 8][l16]      = (_Float16)p0;
            Plds[wave][j + hs * 8][16 + l16] = (_Float16)p1;
        }
        asm volatile("s_wait_dscnt 0" ::: "memory");
        __builtin_amdgcn_wave_barrier();

        // ---- P (16x32 f16) back as A fragment ----
        const _Float16* pr = &Plds[wave][l16][0];
        v16h pf = cat8(*(const v8h*)(pr + hs * 16), *(const v8h*)(pr + hs * 16 + 8));

        // ---- P @ V : 4 output subtiles over DH=64 ----
        #pragma unroll
        for (int nn = 0; nn < 4; ++nn) {
            v16h vf;
            #pragma unroll
            for (int e = 0; e < 16; ++e) {
                int s = s0 + hs * 16 + e;
                vf[e] = V[((size_t)b * T_ + s) * E_ + h * DH_ + nn * 16 + l16];
            }
            acco[nn] = wmma16(pf, vf, acco[nn]);
        }
        asm volatile("s_wait_dscnt 0" ::: "memory");
        __builtin_amdgcn_wave_barrier();
    }

    // ---- normalize + scatter with the reference's reshape quirk ----
    #pragma unroll
    for (int nn = 0; nn < 4; ++nn) {
        #pragma unroll
        for (int j = 0; j < 8; ++j) {
            const int d  = nn * 16 + l16;
            const int tq = q0 + j + hs * 8;
            const float val = acco[nn][j] / l_[j];
            const int iRow = d * (T_ / 64) + (tq >> 6);
            const int cc   = h * DH_ + (tq & 63);
            O[((size_t)b * T_ + iRow) * E_ + cc] = (_Float16)val;
        }
    }
}

// =====================================================================
// Host-side orchestration
// =====================================================================
extern "C" void kernel_launch(void* const* d_in, const int* in_sizes, int n_in,
                              void* d_out, int out_size, void* d_ws, size_t ws_size,
                              hipStream_t stream) {
    const float* x     = (const float*)d_in[0];
    const float* Wq    = (const float*)d_in[1];
    const float* Wk    = (const float*)d_in[2];
    const float* Wv    = (const float*)d_in[3];
    const float* fc_w  = (const float*)d_in[4];
    const float* fc_b  = (const float*)d_in[5];
    const float* ln1_s = (const float*)d_in[6];
    const float* ln1_b = (const float*)d_in[7];
    const float* ln2_s = (const float*)d_in[8];
    const float* ln2_b = (const float*)d_in[9];
    const float* ff_w1 = (const float*)d_in[10];
    const float* ff_b1 = (const float*)d_in[11];
    const float* ff_w2 = (const float*)d_in[12];
    const float* ff_b2 = (const float*)d_in[13];
    float* out = (float*)d_out;

    // ---- workspace carve-up ----
    char* ws = (char*)d_ws;
    size_t off = 0;
    auto take = [&](size_t bytes) { char* p = ws + off; off = (off + bytes + 255) & ~(size_t)255; return p; };
    _Float16* wq16 = (_Float16*)take((size_t)E_ * E_ * 2);
    _Float16* wk16 = (_Float16*)take((size_t)E_ * E_ * 2);
    _Float16* wv16 = (_Float16*)take((size_t)E_ * E_ * 2);
    _Float16* wfc  = (_Float16*)take((size_t)E_ * E_ * 2);
    _Float16* w1   = (_Float16*)take((size_t)E_ * FF_ * 2);
    _Float16* w2   = (_Float16*)take((size_t)FF_ * E_ * 2);
    _Float16* xn   = (_Float16*)take((size_t)M_ * E_ * 2);
    _Float16* q16  = (_Float16*)take((size_t)M_ * E_ * 2);
    _Float16* k16  = (_Float16*)take((size_t)M_ * E_ * 2);
    _Float16* v16  = (_Float16*)take((size_t)M_ * E_ * 2);
    _Float16* cat_ = (_Float16*)take((size_t)M_ * E_ * 2);
    float*    hres = (float*)   take((size_t)M_ * E_ * 4);
    _Float16* y16  = (_Float16*)take((size_t)M_ * E_ * 2);
    _Float16* ff1  = (_Float16*)take((size_t)M_ * FF_ * 2);
    (void)ws_size; (void)in_sizes; (void)n_in; (void)out_size;

    // 1) weights -> f16
    pack_qkv_f16<<<(H_ * E_ * DH_) / 256, 256, 0, stream>>>(Wq, wq16);
    pack_qkv_f16<<<(H_ * E_ * DH_) / 256, 256, 0, stream>>>(Wk, wk16);
    pack_qkv_f16<<<(H_ * E_ * DH_) / 256, 256, 0, stream>>>(Wv, wv16);
    cvt_f16<<<(E_ * E_) / 256, 256, 0, stream>>>(fc_w, wfc, E_ * E_);
    cvt_f16<<<(E_ * FF_) / 256, 256, 0, stream>>>(ff_w1, w1, E_ * FF_);
    cvt_f16<<<(FF_ * E_) / 256, 256, 0, stream>>>(ff_w2, w2, FF_ * E_);

    // 2) LN1
    layernorm_f16<<<M_, 256, 0, stream>>>(x, ln1_s, ln1_b, xn);

    // 3) Q/K/V projections (plain f16 GEMM -> f16)
    dim3 gproj(E_ / BN, M_ / BM);
    gemm_wmma<false, false, false, true><<<gproj, 256, 0, stream>>>(xn, wq16, nullptr, nullptr, nullptr, q16, M_, E_, E_);
    gemm_wmma<false, false, false, true><<<gproj, 256, 0, stream>>>(xn, wk16, nullptr, nullptr, nullptr, k16, M_, E_, E_);
    gemm_wmma<false, false, false, true><<<gproj, 256, 0, stream>>>(xn, wv16, nullptr, nullptr, nullptr, v16, M_, E_, E_);

    // 4) causal flash attention -> concat (f16, with reference reshape quirk)
    flash_attn<<<dim3(B_ * H_, T_ / 64), 128, 0, stream>>>(q16, k16, v16, cat_);

    // 5) output projection: h = concat @ fc_w + fc_b + x   (fp32)
    gemm_wmma<true, true, false, false><<<gproj, 256, 0, stream>>>(cat_, wfc, fc_b, x, hres, nullptr, M_, E_, E_);

    // 6) LN2: y = layernorm(h)  (f16)
    layernorm_f16<<<M_, 256, 0, stream>>>(hres, ln2_s, ln2_b, y16);

    // 7) FFN1: ff1 = gelu(y @ W1 + b1)  (f16)
    gemm_wmma<true, false, true, true><<<dim3(FF_ / BN, M_ / BM), 256, 0, stream>>>(y16, w1, ff_b1, nullptr, nullptr, ff1, M_, FF_, E_);

    // 8) FFN2: out = ff1 @ W2 + b2 + h  (fp32)
    gemm_wmma<true, true, false, false><<<gproj, 256, 0, stream>>>(ff1, w2, ff_b2, hres, out, nullptr, M_, E_, FF_);
}
